// MixtureExpertsMlp_4956392259792
// MI455X (gfx1250) — matrix operand order repaired and checked by reading, not compiled, not run on hardware
//
#include <hip/hip_runtime.h>
#include <hip/hip_bf16.h>

typedef __bf16 bf16;
typedef __attribute__((ext_vector_type(16))) __bf16 v16bf;
typedef __attribute__((ext_vector_type(8)))  float  v8f;
typedef int v4i __attribute__((vector_size(16)));

#define TILE_M 128
#define TILE_N 128
#define TILE_K 32
#define THREADS 256

union FragAB { v16bf v; uint4 q[2]; };

enum { MODE_F32 = 0, MODE_BF16 = 1, MODE_BF16_BIAS_GELU = 2, MODE_BF16_BIAS_T = 3 };

// ---- CDNA5 async global->LDS copy helpers ---------------------------------
#if __has_builtin(__builtin_amdgcn_global_load_async_to_lds_b128)
#define HAVE_ASYNC_LDS 1
#else
#define HAVE_ASYNC_LDS 0
#endif

// generic->AS1 is an identity mapping for global pointers; generic LDS
// addresses carry the LDS byte offset in addr[31:0] (ISA aperture rules),
// so integer-casting to a 32-bit AS3 pointer yields the LDS offset.
#define AS1V4(p) ((__attribute__((address_space(1))) v4i*)(unsigned long long)(p))
#define AS3V4(p) ((__attribute__((address_space(3))) v4i*)(unsigned int)(unsigned long long)(p))

__device__ __forceinline__ void async_copy16(const bf16* g, bf16* l) {
#if HAVE_ASYNC_LDS
  __builtin_amdgcn_global_load_async_to_lds_b128(AS1V4(g), AS3V4(l), 0, 0);
#else
  *(uint4*)l = *(const uint4*)g;   // sync fallback: load + ds_store
#endif
}

__device__ __forceinline__ void wait_async0() {
#if HAVE_ASYNC_LDS
#if __has_builtin(__builtin_amdgcn_s_wait_asynccnt)
  __builtin_amdgcn_s_wait_asynccnt(0);
#else
  asm volatile("s_wait_asynccnt 0x0" ::: "memory");
#endif
#endif
}

__device__ __forceinline__ float gelu_tanh(float x) {
  const float c0 = 0.7978845608028654f;  // sqrt(2/pi)
  const float c1 = 0.044715f;
  float x3 = x * x * x;
  return 0.5f * x * (1.0f + tanhf(c0 * (x + c1 * x3)));
}

// ---------------------------------------------------------------------------
// Generic bf16 WMMA GEMM: C[z] = A[z] (MxK, row-major) * B[z]^T (NcolxK, row-major)
// Block tile 128x128, K-step 32, 8 waves (wave32) each owning a 64x32 sub-tile
// = 4x2 fragments of v_wmma_f32_16x16x32_bf16. LDS double buffered; tiles are
// staged with GLOBAL_LOAD_ASYNC_TO_LDS_B128 (no staging VGPRs -> no spills).
// Per-z offsets: off = (z/zdiv)*outer + (z%zdiv)*inner.
// ---------------------------------------------------------------------------
template<int MODE>
__global__ __launch_bounds__(THREADS)
void k_gemm_wmma(const bf16* __restrict__ A, const bf16* __restrict__ Bm,
                 void* __restrict__ Cv, const float* __restrict__ bias,
                 int M, int Ncol, int K, int zdiv,
                 long long sA_outer, long long sA_inner,
                 long long sB_outer, long long sB_inner,
                 long long sC_outer, long long sC_inner,
                 long long sBias_inner, int ldC)
{
  __shared__ bf16 ldsA[2][TILE_M][TILE_K];
  __shared__ bf16 ldsB[2][TILE_N][TILE_K];

  const int tid  = threadIdx.x;
  const int lane = tid & 31;
  const int w    = tid >> 5;
  const int wm   = (w & 1) * 64;        // wave M origin within tile
  const int wn   = (w >> 1) * 32;       // wave N origin within tile
  const int lr   = lane & 15;
  const int hi   = lane >> 4;

  const int z = blockIdx.z;
  const long long Aoff = (long long)(z / zdiv) * sA_outer + (long long)(z % zdiv) * sA_inner;
  const long long Boff = (long long)(z / zdiv) * sB_outer + (long long)(z % zdiv) * sB_inner;
  const bf16* gA = A  + Aoff + (size_t)blockIdx.y * TILE_M * (size_t)K;
  const bf16* gB = Bm + Boff + (size_t)blockIdx.x * TILE_N * (size_t)K;

  // per-thread tile chunks: c = tid + i*256 -> row c>>2, 16B chunk c&3
  const int r0 = tid >> 2, ch0 = (tid & 3) * 8;
  const int r1 = (tid + THREADS) >> 2, ch1 = ((tid + THREADS) & 3) * 8;

  v8f acc[4][2];
#pragma unroll
  for (int mi = 0; mi < 4; ++mi)
#pragma unroll
    for (int ni = 0; ni < 2; ++ni)
      acc[mi][ni] = (v8f){0.f, 0.f, 0.f, 0.f, 0.f, 0.f, 0.f, 0.f};

  const int nkt = K / TILE_K;

  // prologue: stage k-tile 0 into buffer 0
  async_copy16(gA + (size_t)r0 * K + ch0, &ldsA[0][r0][ch0]);
  async_copy16(gB + (size_t)r0 * K + ch0, &ldsB[0][r0][ch0]);
  async_copy16(gA + (size_t)r1 * K + ch1, &ldsA[0][r1][ch1]);
  async_copy16(gB + (size_t)r1 * K + ch1, &ldsB[0][r1][ch1]);
  wait_async0();
  __syncthreads();

  for (int t = 0; t < nkt; ++t) {
    const int cur = t & 1;
    const int nxt = cur ^ 1;
    const bool have_next = (t + 1) < nkt;

    if (have_next) {   // DMA next tile straight into the other LDS buffer
      const int kn = (t + 1) * TILE_K;
      async_copy16(gA + (size_t)r0 * K + kn + ch0, &ldsA[nxt][r0][ch0]);
      async_copy16(gB + (size_t)r0 * K + kn + ch0, &ldsB[nxt][r0][ch0]);
      async_copy16(gA + (size_t)r1 * K + kn + ch1, &ldsA[nxt][r1][ch1]);
      async_copy16(gB + (size_t)r1 * K + kn + ch1, &ldsB[nxt][r1][ch1]);
    }

    // Fragment loads from LDS per documented wave32 VGPR layouts.
    FragAB fa[4], fb[2];
#pragma unroll
    for (int mi = 0; mi < 4; ++mi) {
      int rrow = wm + mi * 16 + lr;            // A: M = lane%16, K-half by lane>=16
      fa[mi].q[0] = *(const uint4*)&ldsA[cur][rrow][hi * 8];
      fa[mi].q[1] = *(const uint4*)&ldsA[cur][rrow][hi * 8 + 16];
    }
#pragma unroll
    for (int ni = 0; ni < 2; ++ni) {
      int col = wn + ni * 16 + lr;             // B: N = lane%16, K-half by lane>=16
      fb[ni].q[0] = *(const uint4*)&ldsB[cur][col][hi * 8];
      fb[ni].q[1] = *(const uint4*)&ldsB[cur][col][hi * 8 + 16];
    }

#pragma unroll
    for (int mi = 0; mi < 4; ++mi)
#pragma unroll
      for (int ni = 0; ni < 2; ++ni)
        acc[mi][ni] = __builtin_amdgcn_wmma_f32_16x16x32_bf16(
            false, fa[mi].v, false, fb[ni].v, (short)0, acc[mi][ni], false, false);

    if (have_next) wait_async0();
    __syncthreads();
  }

  // Epilogue. C fragment: VGPR r -> M = r + 8*(lane>=16), N = lane%16.
  const long long Coff    = (long long)(z / zdiv) * sC_outer + (long long)(z % zdiv) * sC_inner;
  const long long biasOff = (long long)(z % zdiv) * sBias_inner;
  const int gm0 = blockIdx.y * TILE_M + wm + hi * 8;
  const int gn0 = blockIdx.x * TILE_N + wn + lr;

#pragma unroll
  for (int mi = 0; mi < 4; ++mi)
#pragma unroll
    for (int ni = 0; ni < 2; ++ni)
#pragma unroll
      for (int r = 0; r < 8; ++r) {
        const int gm = gm0 + mi * 16 + r;
        const int gn = gn0 + ni * 16;
        float v = acc[mi][ni][r];
        if (MODE == MODE_F32) {
          ((float*)Cv)[Coff + (long long)gm * ldC + gn] = v;
        } else if (MODE == MODE_BF16) {
          ((bf16*)Cv)[Coff + (long long)gm * ldC + gn] = (bf16)v;
        } else if (MODE == MODE_BF16_BIAS_GELU) {
          v += bias[biasOff + gn];
          ((bf16*)Cv)[Coff + (long long)gm * ldC + gn] = (bf16)gelu_tanh(v);
        } else {  // MODE_BF16_BIAS_T : store transposed (column index gm contiguous)
          v += bias[biasOff + gn];
          ((bf16*)Cv)[Coff + (long long)gn * ldC + gm] = (bf16)v;
        }
      }
}

// ---------------------------------------------------------------------------
// fp32 -> bf16 convert (flat) and transpose-convert src[z][r][c] -> dst[z][c][r]
// ---------------------------------------------------------------------------
__global__ void k_cvt(const float* __restrict__ s, bf16* __restrict__ d, size_t n) {
  size_t i = (size_t)blockIdx.x * THREADS + threadIdx.x;
  if (i < n) d[i] = (bf16)s[i];
}

__global__ void k_cvt_T(const float* __restrict__ s, bf16* __restrict__ d, int R, int C) {
  const size_t plane = (size_t)R * C;
  const size_t base = (size_t)blockIdx.y * plane;
  const size_t i = (size_t)blockIdx.x * THREADS + threadIdx.x;
  const int r = (int)(i / C), c = (int)(i % C);
  d[base + (size_t)c * R + r] = (bf16)s[base + i];
}

// ---------------------------------------------------------------------------
// Dispatch softmax: one block per (b,e,s) row of logitsT (length n=4096),
// output bf16 dispatchT with the same layout.
// ---------------------------------------------------------------------------
__global__ __launch_bounds__(THREADS)
void k_row_softmax(const float* __restrict__ src, bf16* __restrict__ dst, int n) {
  __shared__ float red[THREADS];
  const int tid = threadIdx.x;
  const size_t row = blockIdx.x;
  const float* r = src + row * (size_t)n;
  float v[16];
  float m = -3.4e38f;
#pragma unroll
  for (int i = 0; i < 16; ++i) { v[i] = r[tid + (i << 8)]; m = fmaxf(m, v[i]); }
  red[tid] = m; __syncthreads();
  for (int s2 = 128; s2 > 0; s2 >>= 1) {
    if (tid < s2) red[tid] = fmaxf(red[tid], red[tid + s2]);
    __syncthreads();
  }
  m = red[0]; __syncthreads();
  float s = 0.f;
#pragma unroll
  for (int i = 0; i < 16; ++i) { v[i] = __expf(v[i] - m); s += v[i]; }
  red[tid] = s; __syncthreads();
  for (int s2 = 128; s2 > 0; s2 >>= 1) {
    if (tid < s2) red[tid] += red[tid + s2];
    __syncthreads();
  }
  const float inv = 1.f / red[0];
  bf16* o = dst + row * (size_t)n;
#pragma unroll
  for (int i = 0; i < 16; ++i) o[tid + (i << 8)] = (bf16)(v[i] * inv);
}

// ---------------------------------------------------------------------------
// Combine softmax: softmax over all E*S slots for each (b,n), reading
// logitsT[b][es][n] columns (coalesced 64-token strips), online max/sum,
// writes combine[b][n][es] (bf16, es-contiguous for the output GEMM).
// ---------------------------------------------------------------------------
__global__ __launch_bounds__(THREADS)
void k_combine_softmax(const float* __restrict__ logitsT, bf16* __restrict__ comb,
                       int N, int ES) {
  const int bpb = N / 64;
  const int b = blockIdx.x / bpb;
  const int n0 = (blockIdx.x % bpb) * 64;
  const int tok = threadIdx.x & 63;
  const int q = threadIdx.x >> 6;
  const float* base = logitsT + (size_t)b * ES * N + n0 + tok;
  float m = -3.4e38f, s = 0.f;
  for (int es = q; es < ES; es += 4) {
    float v = base[(size_t)es * N];
    float mn = fmaxf(m, v);
    s = s * __expf(m - mn) + __expf(v - mn);
    m = mn;
  }
  __shared__ float Ms[4][64], Ss[4][64], Mf[64], Sf[64];
  Ms[q][tok] = m; Ss[q][tok] = s;
  __syncthreads();
  if (q == 0) {
    float mm = Ms[0][tok], ss = Ss[0][tok];
#pragma unroll
    for (int j = 1; j < 4; ++j) {
      float mj = Ms[j][tok];
      float mn = fmaxf(mm, mj);
      ss = ss * __expf(mm - mn) + Ss[j][tok] * __expf(mj - mn);
      mm = mn;
    }
    Mf[tok] = mm; Sf[tok] = ss;
  }
  __syncthreads();
  const float M = Mf[tok];
  const float inv = 1.f / Sf[tok];
  bf16* o = comb + ((size_t)b * N + n0 + tok) * ES;
  for (int es = q; es < ES; es += 4)
    o[es] = (bf16)(__expf(base[(size_t)es * N] - M) * inv);
}

// ---------------------------------------------------------------------------
extern "C" void kernel_launch(void* const* d_in, const int* in_sizes, int n_in,
                              void* d_out, int out_size, void* d_ws, size_t ws_size,
                              hipStream_t stream) {
  const int B = 4, Ntok = 4096, D = 768, E = 4, S = 1024, H = 3072;
  const int ES = E * S;

  const float* x   = (const float*)d_in[0];
  const float* phi = (const float*)d_in[1];
  const float* w1  = (const float*)d_in[2];
  const float* b1  = (const float*)d_in[3];
  const float* w2  = (const float*)d_in[4];
  const float* b2  = (const float*)d_in[5];
  float* out = (float*)d_out;

  char* p = (char*)d_ws;
  auto take = [&](size_t bytes) -> char* {
    char* r = p;
    p += (bytes + 255) & ~(size_t)255;
    return r;
  };
  bf16*  xb      = (bf16*)take((size_t)B * Ntok * D * 2);
  bf16*  xT      = (bf16*)take((size_t)B * D * Ntok * 2);
  bf16*  phib    = (bf16*)take((size_t)E * S * D * 2);
  bf16*  w1T     = (bf16*)take((size_t)E * H * D * 2);
  bf16*  w2T     = (bf16*)take((size_t)E * D * H * 2);
  float* logitsT = (float*)take((size_t)B * ES * Ntok * 4);
  bf16*  dispT   = (bf16*)take((size_t)B * ES * Ntok * 2);
  bf16*  comb    = (bf16*)take((size_t)B * Ntok * ES * 2);
  bf16*  slots   = (bf16*)take((size_t)B * E * S * D * 2);
  bf16*  hbuf    = (bf16*)take((size_t)B * E * S * H * 2);
  bf16*  yT      = (bf16*)take((size_t)B * D * ES * 2);

  // 1) bf16 conversions (+ transposed copies for K-contiguous GEMM operands)
  k_cvt<<<(B * Ntok * D) / THREADS, THREADS, 0, stream>>>(x, xb, (size_t)B * Ntok * D);
  k_cvt<<<(E * S * D) / THREADS, THREADS, 0, stream>>>(phi, phib, (size_t)E * S * D);
  k_cvt_T<<<dim3((Ntok * D) / THREADS, B), THREADS, 0, stream>>>(x, xT, Ntok, D);
  k_cvt_T<<<dim3((D * H) / THREADS, E), THREADS, 0, stream>>>(w1, w1T, D, H);
  k_cvt_T<<<dim3((H * D) / THREADS, E), THREADS, 0, stream>>>(w2, w2T, H, D);

  // 2) logitsT[b,e,s,n] = phi[e] (SxD) * x[b]^T        M=S Ncol=N K=D
  k_gemm_wmma<MODE_F32><<<dim3(Ntok / 128, S / 128, B * E), THREADS, 0, stream>>>(
      phib, xb, logitsT, nullptr, S, Ntok, D, E,
      0LL, (long long)S * D,
      (long long)Ntok * D, 0LL,
      (long long)ES * Ntok, (long long)S * Ntok,
      0LL, Ntok);

  // 3) softmaxes
  k_row_softmax<<<B * ES, THREADS, 0, stream>>>(logitsT, dispT, Ntok);
  k_combine_softmax<<<B * (Ntok / 64), THREADS, 0, stream>>>(logitsT, comb, Ntok, ES);

  // 4) slots[b,e,s,d] = dispatchT[b,e] (SxN) * xT[b]^T   M=S Ncol=D K=N
  k_gemm_wmma<MODE_BF16><<<dim3(D / 128, S / 128, B * E), THREADS, 0, stream>>>(
      dispT, xT, slots, nullptr, S, D, Ntok, E,
      (long long)E * S * Ntok, (long long)S * Ntok,
      (long long)D * Ntok, 0LL,
      (long long)E * S * D, (long long)S * D,
      0LL, D);

  // 5) h = gelu(slots * w1T^T + b1)                      M=S Ncol=H K=D
  k_gemm_wmma<MODE_BF16_BIAS_GELU><<<dim3(H / 128, S / 128, B * E), THREADS, 0, stream>>>(
      slots, w1T, hbuf, b1, S, H, D, E,
      (long long)E * S * D, (long long)S * D,
      0LL, (long long)H * D,
      (long long)E * S * H, (long long)S * H,
      (long long)H, H);

  // 6) yT[b,d,e*S+s] = (h * w2T^T + b2)^T                M=S Ncol=D K=H
  k_gemm_wmma<MODE_BF16_BIAS_T><<<dim3(D / 128, S / 128, B * E), THREADS, 0, stream>>>(
      hbuf, w2T, yT, b2, S, D, H, E,
      (long long)E * S * H, (long long)S * H,
      0LL, (long long)D * H,
      (long long)D * ES, (long long)S,
      (long long)D, ES);

  // 7) out[b,n,d] = combine[b] (NxES) * yT[b]^T          M=N Ncol=D K=ES
  k_gemm_wmma<MODE_F32><<<dim3(D / 128, Ntok / 128, B), THREADS, 0, stream>>>(
      comb, yT, out, nullptr, Ntok, D, ES, 1,
      (long long)Ntok * ES, 0LL,
      (long long)D * ES, 0LL,
      (long long)Ntok * D, 0LL,
      0LL, D);
}
